// StandardMHA_89154931130564
// MI455X (gfx1250) — matrix operand order repaired and checked by reading, not compiled, not run on hardware
//
#include <hip/hip_runtime.h>
#include <stdint.h>

#define SEQ   4096
#define DIM   2048
#define NHEAD 16
#define HDIM  128

typedef __attribute__((ext_vector_type(16))) __bf16 v16bf;
typedef __attribute__((ext_vector_type(8)))  float  v8f;

union Frag {
    uint32_t u[8];
    v16bf    v;
};

__device__ __forceinline__ unsigned short f2bf(float f) {
    union { float f; uint32_t u; } x; x.f = f;
    uint32_t u = x.u;
    u += 0x7FFFu + ((u >> 16) & 1u);   // round-to-nearest-even
    return (unsigned short)(u >> 16);
}

// A-matrix (16x32 bf16) per-lane uint index within a 16-uint (32 bf16) K-chunk:
// half 0: K = {0..7, 16..23}; half 1: K = {8..15, 24..31}; 2 bf16 per VGPR.
__device__ __forceinline__ int uidxA(int v, int half) {
    return (v < 4 ? v : v + 4) + 4 * half;
}
// B-matrix (32x16 bf16): same K order as A: half 0: K = {0..7, 16..23}; half 1: K = {8..15, 24..31}.
__device__ __forceinline__ int uidxB(int v, int half) {
    return (v < 4 ? v : v + 4) + 4 * half;
}

__device__ __forceinline__ v8f wmma_bf16(const Frag& a, const Frag& b, v8f c) {
    return __builtin_amdgcn_wmma_f32_16x16x32_bf16(
        false, a.v, false, b.v, (short)0, c, false, false);
}

// ---- CDNA5 async global->LDS copy (raw bytes, tracked by ASYNCcnt) ----
__device__ __forceinline__ uint32_t lds_off(const void* p) {
    return (uint32_t)(uintptr_t)p;   // low 32 bits of generic addr == LDS offset
}
__device__ __forceinline__ void async_b64(uint32_t ldsoff, unsigned long long gaddr) {
    asm volatile("global_load_async_to_lds_b64 %0, %1, off"
                 :: "v"(ldsoff), "v"(gaddr) : "memory");
}
__device__ __forceinline__ void async_wait0() {
    asm volatile("s_wait_asynccnt 0x0" ::: "memory");
}

enum { OUT_F32 = 0, OUT_BF16_HEADS = 1, OUT_BF16_HEADS_T = 2 };

// Y[m][n] = sum_k A[m][k] * W[n][k] + bias[n]; M=SEQ, N=K=DIM.
// 256 threads = 8 waves; block tile 128(M) x 128(N); wave tile 32x64; K-step 32.
template<int OMODE, bool ABF16>
__global__ __launch_bounds__(256)
void gemm256(const void* __restrict__ Aptr, const float* __restrict__ W,
             const float* __restrict__ bias, void* __restrict__ outp, float oscale)
{
    __shared__ uint32_t Asub[128][18];   // 128 rows x 32 bf16 (16 uints) + pad
    __shared__ uint32_t Bsub[128][18];   // B stored as [n][k]

    const int t    = threadIdx.x;
    const int wid  = t >> 5;
    const int lane = t & 31;
    const int l15  = lane & 15;
    const int half = lane >> 4;
    const int waveM = wid >> 1;          // 0..3
    const int waveN = wid & 1;           // 0..1
    const int m0 = blockIdx.y * 128;
    const int n0 = blockIdx.x * 128;

    v8f acc[2][4] = {};

    for (int k0 = 0; k0 < DIM; k0 += 32) {
        __syncthreads();
        // ---- stage A tile ----
        {
            const int r  = t >> 1;
            const int c0 = (t & 1) * 16;     // f32/bf16-element offset within 32-chunk
            if (!ABF16) {
                const float* A = (const float*)Aptr;
                const float4* src = (const float4*)(A + (size_t)(m0 + r) * DIM + k0 + c0);
                #pragma unroll
                for (int i = 0; i < 4; ++i) {
                    float4 f = src[i];
                    Asub[r][(c0 >> 1) + i * 2 + 0] =
                        (uint32_t)f2bf(f.x) | ((uint32_t)f2bf(f.y) << 16);
                    Asub[r][(c0 >> 1) + i * 2 + 1] =
                        (uint32_t)f2bf(f.z) | ((uint32_t)f2bf(f.w) << 16);
                }
            } else {
                // A already bf16: async DMA straight into LDS (4 x b64 per thread)
                const unsigned long long g =
                    (unsigned long long)(uintptr_t)((const uint32_t*)Aptr +
                        ((size_t)(m0 + r) * DIM + k0 + c0) / 2);
                #pragma unroll
                for (int i = 0; i < 4; ++i)
                    async_b64(lds_off(&Asub[r][(c0 >> 1) + 2 * i]), g + 8ull * i);
            }
        }
        // ---- stage B tile: B[k][n] = W[n][k], stored [n][k] ----
        {
            const int r  = t >> 1;
            const int c0 = (t & 1) * 16;
            const float4* src = (const float4*)(W + (size_t)(n0 + r) * DIM + k0 + c0);
            #pragma unroll
            for (int i = 0; i < 4; ++i) {
                float4 f = src[i];
                Bsub[r][(c0 >> 1) + i * 2 + 0] =
                    (uint32_t)f2bf(f.x) | ((uint32_t)f2bf(f.y) << 16);
                Bsub[r][(c0 >> 1) + i * 2 + 1] =
                    (uint32_t)f2bf(f.z) | ((uint32_t)f2bf(f.w) << 16);
            }
        }
        if (ABF16) async_wait0();
        __syncthreads();

        // ---- fragments + 8 WMMAs ----
        Frag a[2], b[4];
        #pragma unroll
        for (int s = 0; s < 2; ++s) {
            const int row = waveM * 32 + s * 16 + l15;
            #pragma unroll
            for (int v = 0; v < 8; ++v) a[s].u[v] = Asub[row][uidxA(v, half)];
        }
        #pragma unroll
        for (int j = 0; j < 4; ++j) {
            const int col = waveN * 64 + j * 16 + l15;
            #pragma unroll
            for (int v = 0; v < 8; ++v) b[j].u[v] = Bsub[col][uidxB(v, half)];
        }
        #pragma unroll
        for (int i = 0; i < 2; ++i)
            #pragma unroll
            for (int j = 0; j < 4; ++j)
                acc[i][j] = wmma_bf16(a[i], b[j], acc[i][j]);
    }

    // ---- epilogue: bias add + store in requested layout ----
    #pragma unroll
    for (int i = 0; i < 2; ++i) {
        #pragma unroll
        for (int j = 0; j < 4; ++j) {
            const int col = n0 + waveN * 64 + j * 16 + l15;
            const float bv = bias[col];
            #pragma unroll
            for (int r = 0; r < 8; ++r) {
                const int row = m0 + waveM * 32 + i * 16 + r + 8 * half;
                const float val = (acc[i][j][r] + bv) * oscale;
                if (OMODE == OUT_F32) {
                    ((float*)outp)[(size_t)row * DIM + col] = val;
                } else if (OMODE == OUT_BF16_HEADS) {
                    const int h = col >> 7, d = col & 127;
                    ((unsigned short*)outp)[((size_t)h * SEQ + row) * HDIM + d] = f2bf(val);
                } else { // OUT_BF16_HEADS_T: Vt[h][d][s], col == h*128+d
                    ((unsigned short*)outp)[(size_t)col * SEQ + row] = f2bf(val);
                }
            }
        }
    }
}

// Flash attention: grid (32 qblocks, 16 heads), 256 threads = 8 waves.
// K/V tiles for each 32-key step are staged into LDS once per workgroup via
// async global->LDS DMA; each wave owns 16 query rows with online softmax.
__global__ __launch_bounds__(256)
void attn256(const unsigned short* __restrict__ Qb,   // [H][S][Hd] bf16 (pre-scaled)
             const unsigned short* __restrict__ Kb,   // [H][S][Hd] bf16
             const unsigned short* __restrict__ Vt,   // [H][Hd][S] bf16
             unsigned short* __restrict__ Ob)         // [S][DIM]   bf16
{
    __shared__ uint32_t Ks[32][66];                   // [key][d-pair], 8B-aligned rows
    __shared__ uint32_t Vs[128][18];                  // [d][key-pair]
    __shared__ unsigned short Pbuf[8][16][34];        // per-wave 16x32 bf16 P tile

    const int t    = threadIdx.x;
    const int wid  = t >> 5;
    const int lane = t & 31;
    const int l15  = lane & 15;
    const int half = lane >> 4;
    const int h    = blockIdx.y;
    const int q0   = blockIdx.x * 128 + wid * 16;

    const uint32_t* Qu = (const uint32_t*)Qb;
    const uint32_t* Ku = (const uint32_t*)Kb;
    const uint32_t* Vu = (const uint32_t*)Vt;

    // ---- preload Q fragments (16 rows x 128 d = 4 K-steps), kept in VGPRs ----
    Frag qf[4];
    {
        const size_t base = ((size_t)h * SEQ + q0 + l15) * (HDIM / 2);
        #pragma unroll
        for (int ks = 0; ks < 4; ++ks)
            #pragma unroll
            for (int v = 0; v < 8; ++v)
                qf[ks].u[v] = Qu[base + ks * 16 + uidxA(v, half)];
    }

    float mrun[8], lrun[8];
    #pragma unroll
    for (int r = 0; r < 8; ++r) { mrun[r] = -1e30f; lrun[r] = 0.0f; }
    v8f oacc[8] = {};

    for (int kb = 0; kb < SEQ; kb += 32) {
        __syncthreads();   // previous iteration's LDS reads complete

        // ---- async stage K tile: 32 keys x 128 d bf16 ----
        {
            const int key = t >> 3;
            const int c   = (t & 7) * 8;
            const unsigned long long g = (unsigned long long)(uintptr_t)
                (Ku + ((size_t)h * SEQ + kb + key) * (HDIM / 2) + c);
            #pragma unroll
            for (int i = 0; i < 4; ++i)
                async_b64(lds_off(&Ks[key][c + 2 * i]), g + 8ull * i);
        }
        // ---- async stage V tile: 128 d x 32 keys bf16 (from Vt) ----
        {
            const int d = t >> 1;
            const int c = (t & 1) * 8;
            const unsigned long long g = (unsigned long long)(uintptr_t)
                (Vu + ((size_t)h * HDIM + d) * (SEQ / 2) + (kb >> 1) + c);
            #pragma unroll
            for (int i = 0; i < 4; ++i)
                async_b64(lds_off(&Vs[d][c + 2 * i]), g + 8ull * i);
        }
        async_wait0();
        __syncthreads();   // staged tiles visible to all waves

        // ---- scores S = Q K^T for 32 keys (2 tiles of 16) ----
        v8f s0 = {}, s1 = {};
        #pragma unroll
        for (int ks = 0; ks < 4; ++ks) {
            Frag k0f, k1f;
            #pragma unroll
            for (int v = 0; v < 8; ++v) {
                const int ui = ks * 16 + uidxB(v, half);
                k0f.u[v] = Ks[l15][ui];
                k1f.u[v] = Ks[16 + l15][ui];
            }
            s0 = wmma_bf16(qf[ks], k0f, s0);
            s1 = wmma_bf16(qf[ks], k1f, s1);
        }

        // ---- online softmax (rows live across 16 lanes of each half) ----
        float corr[8];
        #pragma unroll
        for (int r = 0; r < 8; ++r) {
            float mx = fmaxf(s0[r], s1[r]);
            #pragma unroll
            for (int d = 1; d < 16; d <<= 1)
                mx = fmaxf(mx, __shfl_xor(mx, d, 32));
            const float mnew = fmaxf(mrun[r], mx);
            const float c  = __expf(mrun[r] - mnew);
            const float p0 = __expf(s0[r] - mnew);
            const float p1 = __expf(s1[r] - mnew);
            float ps = p0 + p1;
            #pragma unroll
            for (int d = 1; d < 16; d <<= 1)
                ps += __shfl_xor(ps, d, 32);
            lrun[r] = lrun[r] * c + ps;
            mrun[r] = mnew;
            corr[r] = c;
            const int m = r + 8 * half;
            Pbuf[wid][m][l15]      = f2bf(p0);
            Pbuf[wid][m][16 + l15] = f2bf(p1);
        }
        #pragma unroll
        for (int tt = 0; tt < 8; ++tt)
            #pragma unroll
            for (int r = 0; r < 8; ++r)
                oacc[tt][r] *= corr[r];

        // make the cross-lane LDS P tile visible (split-counter wait)
        asm volatile("s_wait_dscnt 0x0" ::: "memory");

        // ---- re-layout P (C layout -> A fragment) via LDS ----
        Frag pf;
        #pragma unroll
        for (int v = 0; v < 8; ++v) {
            const int ui = uidxA(v, half);
            pf.u[v] = *(const uint32_t*)&Pbuf[wid][l15][2 * ui];
        }

        // ---- O += P * V from the shared V tile ----
        #pragma unroll
        for (int tt = 0; tt < 8; ++tt) {
            Frag vf;
            const int d = tt * 16 + l15;
            #pragma unroll
            for (int v = 0; v < 8; ++v)
                vf.u[v] = Vs[d][uidxB(v, half)];
            oacc[tt] = wmma_bf16(pf, vf, oacc[tt]);
        }
    }

    // ---- normalize and store O[s][h*128+d] as bf16 ----
    float inv[8];
    #pragma unroll
    for (int r = 0; r < 8; ++r) inv[r] = 1.0f / lrun[r];
    #pragma unroll
    for (int tt = 0; tt < 8; ++tt)
        #pragma unroll
        for (int r = 0; r < 8; ++r) {
            const int row = q0 + r + 8 * half;
            const int col = h * HDIM + tt * 16 + l15;
            Ob[(size_t)row * DIM + col] = f2bf(oacc[tt][r] * inv[r]);
        }
}

extern "C" void kernel_launch(void* const* d_in, const int* in_sizes, int n_in,
                              void* d_out, int out_size, void* d_ws, size_t ws_size,
                              hipStream_t stream) {
    const float* x  = (const float*)d_in[0];
    const float* Wq = (const float*)d_in[1];
    const float* bq = (const float*)d_in[2];
    const float* Wk = (const float*)d_in[3];
    const float* bk = (const float*)d_in[4];
    const float* Wv = (const float*)d_in[5];
    const float* bv = (const float*)d_in[6];
    const float* Wo = (const float*)d_in[7];
    const float* bo = (const float*)d_in[8];

    char* ws = (char*)d_ws;
    const size_t MB16 = (size_t)16 * 1024 * 1024;     // 16*4096*128*2 bytes
    unsigned short* Qb = (unsigned short*)(ws);
    unsigned short* Kb = (unsigned short*)(ws + 1 * MB16);
    unsigned short* Vt = (unsigned short*)(ws + 2 * MB16);
    unsigned short* Ob = (unsigned short*)(ws + 3 * MB16);

    const dim3 blk(256);
    const dim3 gridG(DIM / 128, SEQ / 128);           // 16 x 32
    const float qs = 0.08838834764831845f;            // 1/sqrt(128), folded into Q

    gemm256<OUT_BF16_HEADS,  false><<<gridG, blk, 0, stream>>>(x, Wq, bq, Qb, qs);
    gemm256<OUT_BF16_HEADS,  false><<<gridG, blk, 0, stream>>>(x, Wk, bk, Kb, 1.0f);
    gemm256<OUT_BF16_HEADS_T,false><<<gridG, blk, 0, stream>>>(x, Wv, bv, Vt, 1.0f);

    attn256<<<dim3(SEQ / 128, NHEAD), blk, 0, stream>>>(Qb, Kb, Vt, Ob);

    gemm256<OUT_F32, true><<<gridG, blk, 0, stream>>>(Ob, Wo, bo, d_out, 1.0f);
}
